// WordDecoder_3470333575169
// MI455X (gfx1250) — compile-verified, matrix operand order checked by reading, not run on hardware
//
#include <hip/hip_runtime.h>

// ---------------- problem constants ----------------
#define EMBD   80
#define HIDD   300
#define NVOCAB 512
#define TSTEPS 10
#define NBATCH 16384

// ---------------- tiling constants -----------------
#define MROWS  32          // batch rows per workgroup
#define NWAVES 8
#define NT_G   19          // per-gate N tiles (19*16 = 304 >= 300)
#define KC_H   10          // K chunks over hidden (10*32 = 320 >= 300)
#define KC_X   3           // K chunks over embedding (3*32 = 96 >= 80)
#define NT_V   32          // vocab N tiles (32*16 = 512)

#define H_STRIDE  324      // f32 units  (padded 320, +4 for LDS banking)
#define RZ_STRIDE 304      // f32 units
#define X_STRIDE  104      // bf16 units (padded 96, +8 for LDS banking)

// ---------------- workspace layout (bf16 elements) --
#define WHH_OFF  0
#define WHH_E    (3*NT_G*KC_H*512)          // 291840
#define WIH_OFF  (WHH_OFF + WHH_E)
#define WIH_E    (3*NT_G*KC_X*512)          // 87552
#define WOUT_OFF (WIH_OFF + WIH_E)
#define WOUT_E   (NT_V*KC_H*512)            // 163840
#define EMBB_OFF (WOUT_OFF + WOUT_E)
#define EMBB_E   (NVOCAB*96)                // 49152
#define WS_ELEMS (EMBB_OFF + EMBB_E)        // 592384 bf16 (~1.19 MB)

// ---------------- LDS layout (f32 offsets) ----------
#define LDS_H0   0
#define LDS_H1   (MROWS*H_STRIDE)
#define LDS_R    (2*MROWS*H_STRIDE)
#define LDS_Z    (LDS_R + MROWS*RZ_STRIDE)
#define LDS_X    (LDS_Z + MROWS*RZ_STRIDE)          // bf16 region
#define LDS_AM   (LDS_X + (MROWS*X_STRIDE)/2)       // u64[32]
#define LDS_PRED (LDS_AM + 64)
#define LDS_TOTAL (LDS_PRED + 32)                   // 41952 f32 = 167808 B

typedef __attribute__((ext_vector_type(16))) __bf16 v16bf;
typedef __attribute__((ext_vector_type(8)))  float  v8f;

struct __align__(16) F128 { float x, y, z, w; };
struct __align__(16) U128 { unsigned int x, y, z, w; };
union  BFrag { v16bf v; U128 q[2]; };

// ---------------------------------------------------------------------------
// Prep: pack W_hh / W_ih / W_out into bf16 B-fragment order, emb into bf16.
// B fragment (16x16x32 bf16): lane L holds column N = L&15; element e holds
// K = kc*32 + (e&7) + (e>=8 ? 16 : 0) + (L>=16 ? 8 : 0).
// ---------------------------------------------------------------------------
__global__ void wd_prep_kernel(const float* __restrict__ Wih,
                               const float* __restrict__ Whh,
                               const float* __restrict__ Wout,
                               const float* __restrict__ emb,
                               __bf16* __restrict__ ws) {
  int idx = blockIdx.x * 256 + threadIdx.x;
  if (idx >= WS_ELEMS) return;

  if (idx < WHH_E) {                       // W_hh [900][300]
    int blk = idx >> 9, t = idx & 511;
    int g = blk / (NT_G*KC_H), rem = blk % (NT_G*KC_H);
    int nt = rem / KC_H, kc = rem % KC_H;
    int lane = t >> 4, e = t & 15;
    int n = nt*16 + (lane & 15);
    int k = kc*32 + (e & 7) + ((e >> 3) << 4) + ((lane >> 4) << 3);
    float v = (n < HIDD && k < HIDD) ? Whh[(size_t)(g*HIDD + n)*HIDD + k] : 0.f;
    ws[WHH_OFF + idx] = (__bf16)v;
  } else if (idx < WIH_OFF + WIH_E) {      // W_ih [900][80]
    int j = idx - WIH_OFF;
    int blk = j >> 9, t = j & 511;
    int g = blk / (NT_G*KC_X), rem = blk % (NT_G*KC_X);
    int nt = rem / KC_X, kc = rem % KC_X;
    int lane = t >> 4, e = t & 15;
    int n = nt*16 + (lane & 15);
    int k = kc*32 + (e & 7) + ((e >> 3) << 4) + ((lane >> 4) << 3);
    float v = (n < HIDD && k < EMBD) ? Wih[(size_t)(g*HIDD + n)*EMBD + k] : 0.f;
    ws[idx] = (__bf16)v;
  } else if (idx < WOUT_OFF + WOUT_E) {    // W_out [512][300]
    int j = idx - WOUT_OFF;
    int blk = j >> 9, t = j & 511;
    int nt = blk / KC_H, kc = blk % KC_H;
    int lane = t >> 4, e = t & 15;
    int n = nt*16 + (lane & 15);
    int k = kc*32 + (e & 7) + ((e >> 3) << 4) + ((lane >> 4) << 3);
    float v = (k < HIDD) ? Wout[(size_t)n*HIDD + k] : 0.f;
    ws[idx] = (__bf16)v;
  } else {                                 // emb [512][80] -> [512][96] bf16
    int j = idx - EMBB_OFF;
    int row = j / 96, c = j % 96;
    float v = (c < EMBD) ? emb[(size_t)row*EMBD + c] : 0.f;
    ws[idx] = (__bf16)v;
  }
}

// ---------------------------------------------------------------------------
// Fragment loaders
// A fragment (16x16x32 bf16): lane L holds row M = L&15; element e holds
// K = kc*32 + (e&7) + (e>=8 ? 16 : 0) + (L>=16 ? 8 : 0).
// ---------------------------------------------------------------------------
__device__ inline v16bf load_a_f32(const float* hbase, int mt, int kc, int lane) {
  int m  = mt*16 + (lane & 15);
  int kb = kc*32 + ((lane >> 4) << 3);
  const float* pr = hbase + m*H_STRIDE + kb;
  F128 f0 = *(const F128*)(pr);
  F128 f1 = *(const F128*)(pr + 4);
  F128 f2 = *(const F128*)(pr + 16);
  F128 f3 = *(const F128*)(pr + 20);
  v16bf a;
  a[0]=(__bf16)f0.x; a[1]=(__bf16)f0.y; a[2] =(__bf16)f0.z; a[3] =(__bf16)f0.w;
  a[4]=(__bf16)f1.x; a[5]=(__bf16)f1.y; a[6] =(__bf16)f1.z; a[7] =(__bf16)f1.w;
  a[8]=(__bf16)f2.x; a[9]=(__bf16)f2.y; a[10]=(__bf16)f2.z; a[11]=(__bf16)f2.w;
  a[12]=(__bf16)f3.x;a[13]=(__bf16)f3.y;a[14]=(__bf16)f3.z; a[15]=(__bf16)f3.w;
  return a;
}

__device__ inline v16bf load_a_bf16(const __bf16* xbase, int mt, int kc, int lane) {
  int m  = mt*16 + (lane & 15);
  int kb = kc*32 + ((lane >> 4) << 3);
  const __bf16* pr = xbase + m*X_STRIDE + kb;
  BFrag u;
  u.q[0] = *(const U128*)(pr);
  u.q[1] = *(const U128*)(pr + 16);
  return u.v;
}

__device__ inline v16bf load_b_pack(const __bf16* base, int blk, int lane) {
  const U128* pq = (const U128*)(base + (size_t)blk * 512);
  BFrag u;
  u.q[0] = pq[lane*2 + 0];
  u.q[1] = pq[lane*2 + 1];
  return u.v;
}

__device__ inline v8f wmma_bf16(v16bf a, v16bf b, v8f c) {
  return __builtin_amdgcn_wmma_f32_16x16x32_bf16(false, a, false, b,
                                                 (short)0, c, false, false);
}

__device__ inline float sigmoidf_fast(float x) {
  return 1.f / (1.f + __expf(-x));
}

// ---------------------------------------------------------------------------
// Main decoder: one WG = 32 batch rows, all 10 steps resident in LDS.
// ---------------------------------------------------------------------------
__global__ __launch_bounds__(256, 1)
void wd_gru_decode_kernel(const float* __restrict__ hidden,
                          const float* __restrict__ b_ih,
                          const float* __restrict__ b_hh,
                          const float* __restrict__ b_out,
                          const __bf16* __restrict__ ws,
                          float* __restrict__ out) {
  extern __shared__ float smem[];
  float* Rb = smem + LDS_R;
  float* Zb = smem + LDS_Z;
  __bf16* Xb = (__bf16*)(smem + LDS_X);
  unsigned long long* AM = (unsigned long long*)(smem + LDS_AM);
  int* PRED = (int*)(smem + LDS_PRED);

  const int tid    = threadIdx.x;
  const int lane   = tid & 31;
  const int wave   = tid >> 5;
  const int batch0 = blockIdx.x * MROWS;

  const __bf16* Whh  = ws + WHH_OFF;
  const __bf16* Wih  = ws + WIH_OFF;
  const __bf16* Wout = ws + WOUT_OFF;
  const __bf16* Embb = ws + EMBB_OFF;

  // zero all LDS (zero-pads H K-tail, clears argmax slots)
  for (int i = tid; i < LDS_TOTAL; i += 256) smem[i] = 0.f;
  __syncthreads();

  // init h0 and x0 = ones
  for (int i = tid; i < MROWS*HIDD; i += 256) {
    int m = i / HIDD, k = i % HIDD;
    smem[LDS_H0 + m*H_STRIDE + k] = hidden[(size_t)(batch0 + m)*HIDD + k];
  }
  for (int i = tid; i < MROWS*EMBD; i += 256) {
    int m = i / EMBD, c = i % EMBD;
    Xb[m*X_STRIDE + c] = (__bf16)1.0f;
  }
  __syncthreads();

  int p = 0;
  for (int step = 0; step < TSTEPS; ++step) {
    // ping-pong bases computed from integer offsets (no LDS-pointer array!)
    const float* Hc = smem + (p ? LDS_H1 : LDS_H0);
    float*       Hn = smem + (p ? LDS_H0 : LDS_H1);

    // ---- Pass 1: r,z gates: sigmoid(x@Wih_g + h@Whh_g + b) -> LDS --------
    for (int j = wave; j < 2*2*NT_G; j += NWAVES) {
      int g   = j / (2*NT_G);
      int rem = j % (2*NT_G);
      int mt  = rem / NT_G, nt = rem % NT_G;
      v8f acc = {};
#pragma unroll
      for (int kc = 0; kc < KC_X; ++kc)
        acc = wmma_bf16(load_a_bf16(Xb, mt, kc, lane),
                        load_b_pack(Wih, (g*NT_G + nt)*KC_X + kc, lane), acc);
#pragma unroll
      for (int kc = 0; kc < KC_H; ++kc)
        acc = wmma_bf16(load_a_f32(Hc, mt, kc, lane),
                        load_b_pack(Whh, (g*NT_G + nt)*KC_H + kc, lane), acc);
      int ncol = nt*16 + (lane & 15);
      float bias = (ncol < HIDD) ? (b_ih[g*HIDD + ncol] + b_hh[g*HIDD + ncol]) : 0.f;
      float* dst = (g == 0) ? Rb : Zb;
#pragma unroll
      for (int r = 0; r < 8; ++r) {
        int m = mt*16 + r + ((lane >> 4) << 3);
        dst[m*RZ_STRIDE + ncol] = sigmoidf_fast(acc[r] + bias);
      }
    }
    __syncthreads();

    // ---- Pass 2: n gate (separate i/h accs) + GRU update -> Hn -----------
    for (int j = wave; j < 2*NT_G; j += NWAVES) {
      int mt = j / NT_G, nt = j % NT_G;
      v8f ai = {}, ah = {};
#pragma unroll
      for (int kc = 0; kc < KC_X; ++kc)
        ai = wmma_bf16(load_a_bf16(Xb, mt, kc, lane),
                       load_b_pack(Wih, (2*NT_G + nt)*KC_X + kc, lane), ai);
#pragma unroll
      for (int kc = 0; kc < KC_H; ++kc)
        ah = wmma_bf16(load_a_f32(Hc, mt, kc, lane),
                       load_b_pack(Whh, (2*NT_G + nt)*KC_H + kc, lane), ah);
      int ncol = nt*16 + (lane & 15);
      bool valid = ncol < HIDD;
      float bi = valid ? b_ih[2*HIDD + ncol] : 0.f;
      float bh = valid ? b_hh[2*HIDD + ncol] : 0.f;
#pragma unroll
      for (int r = 0; r < 8; ++r) {
        int m = mt*16 + r + ((lane >> 4) << 3);
        float rr = Rb[m*RZ_STRIDE + ncol];
        float zz = Zb[m*RZ_STRIDE + ncol];
        float hp = Hc[m*H_STRIDE + ncol];
        float nn = tanhf(ai[r] + bi + rr * (ah[r] + bh));
        float hv = (1.f - zz) * nn + zz * hp;
        Hn[m*H_STRIDE + ncol] = valid ? hv : 0.f;
      }
    }
    __syncthreads();

    // ---- Pass 3: logits = h_new@Wout^T + b_out; store + argmax -----------
    for (int j = wave; j < 2*NT_V; j += NWAVES) {
      int mt = j >> 5, nt = j & 31;
      v8f acc = {};
#pragma unroll
      for (int kc = 0; kc < KC_H; ++kc)
        acc = wmma_bf16(load_a_f32(Hn, mt, kc, lane),
                        load_b_pack(Wout, nt*KC_H + kc, lane), acc);
      int ncol = nt*16 + (lane & 15);
      float bo = b_out[ncol];
#pragma unroll
      for (int r = 0; r < 8; ++r) {
        int m = mt*16 + r + ((lane >> 4) << 3);
        float v = acc[r] + bo;
        out[((size_t)step*NBATCH + batch0 + m)*NVOCAB + ncol] = v;
        // argmax: monotonic key, first-index tie-break via (511-ncol)
        unsigned u   = __float_as_uint(v);
        unsigned key = (u & 0x80000000u) ? ~u : (u | 0x80000000u);
        unsigned long long enc =
            ((unsigned long long)key << 32) | (unsigned)(NVOCAB - 1 - ncol);
#pragma unroll
        for (int s = 1; s < 16; s <<= 1) {
          unsigned hi = __shfl_xor((unsigned)(enc >> 32), s, 32);
          unsigned lo = __shfl_xor((unsigned)(enc & 0xFFFFFFFFu), s, 32);
          unsigned long long o = ((unsigned long long)hi << 32) | lo;
          if (o > enc) enc = o;
        }
        if ((lane & 15) == 0) atomicMax(&AM[m], enc);
      }
    }
    __syncthreads();

    // ---- Phase 4: decode preds, gather embeddings, reset argmax ----------
    if (tid < MROWS) {
      unsigned long long am = AM[tid];
      PRED[tid] = NVOCAB - 1 - (int)(am & 0xFFFFFFFFull);
      AM[tid] = 0ull;
    }
    __syncthreads();
    for (int i = tid; i < MROWS*96; i += 256) {
      int m = i / 96, c = i % 96;
      Xb[m*X_STRIDE + c] = Embb[(size_t)PRED[m]*96 + c];
    }
    __syncthreads();

    p ^= 1;
  }
}

// ---------------------------------------------------------------------------
extern "C" void kernel_launch(void* const* d_in, const int* in_sizes, int n_in,
                              void* d_out, int out_size, void* d_ws, size_t ws_size,
                              hipStream_t stream) {
  (void)in_sizes; (void)n_in; (void)out_size; (void)ws_size;
  const float* hidden = (const float*)d_in[0];
  const float* W_ih   = (const float*)d_in[1];
  const float* b_ih   = (const float*)d_in[2];
  const float* W_hh   = (const float*)d_in[3];
  const float* b_hh   = (const float*)d_in[4];
  const float* W_out  = (const float*)d_in[5];
  const float* b_out  = (const float*)d_in[6];
  const float* emb    = (const float*)d_in[7];
  float* out = (float*)d_out;
  __bf16* ws = (__bf16*)d_ws;

  int prepBlocks = (WS_ELEMS + 255) / 256;
  wd_prep_kernel<<<prepBlocks, 256, 0, stream>>>(W_ih, W_hh, W_out, emb, ws);

  size_t shbytes = (size_t)LDS_TOTAL * sizeof(float);   // ~167.8 KB
  wd_gru_decode_kernel<<<NBATCH / MROWS, 256, shbytes, stream>>>(
      hidden, b_ih, b_hh, b_out, ws, out);
}
